// Transformer_P_28853590294610
// MI455X (gfx1250) — compile-verified, matrix operand order checked by reading, not run on hardware
//
#include <hip/hip_runtime.h>
#include <hip/hip_bf16.h>

// ---------------- problem constants (from reference setup_inputs) ----------
#define B_      8
#define NTOK    4096
#define CDIM    512
#define NH      8
#define DHD     64
#define MLAND   256
#define LSEG    17          // NP / MLAND
#define NSEQ    4097        // 1 cls + 4096 tokens (64*64 grid, add==0)
#define PADF    255         // front zero-padding inside nystrom_attention
#define NP      4352        // NSEQ + PADF = MLAND * LSEG
#define BH      64          // B_ * NH
#define HHW     64          // spatial H=W for the conv mixer
#define RESK    33

typedef __bf16 bf16;
typedef __attribute__((ext_vector_type(16))) __bf16 v16bf;
typedef __attribute__((ext_vector_type(8)))  __bf16 bf16x8;
typedef __attribute__((ext_vector_type(8)))  float  v8f;
typedef __attribute__((address_space(3)))    bf16   lds_bf16;

// ---------------------------------------------------------------------------
// WMMA GEMM: D[M,N] = scale * A[M,K](bf16,row-major) x B
//   MODE_B = 0 : B given transposed, Bt[N,K] row-major (direct b128 frags)
//   MODE_B = 1 : B[K,N] row-major; 32x16 B tiles staged to wave-private LDS
//                with double-buffered global_load_async_to_lds_b128 +
//                s_wait_asynccnt, fragments gathered from LDS.
// EPI: 0 = store f32   1 = store bf16
//      2 = QKV split epilogue (scatter into per-head q(0.125x)/k/v bf16 bufs)
//      3 = out-proj epilogue: h[b, ip-PADF, col] += acc + bias[col] (ip>=PADF)
// Register blocking: each wave computes BM x BN 16x16 tiles (A frags reused
// BN times, B frags BM times). 4 waves / block.
// Fragment layout per CDNA5 ISA 7.12.2:
//   lane<16 holds K {0..7,16..23}+kk, lane>=16 holds K {8..15,24..31}+kk.
// ---------------------------------------------------------------------------
template<int MODE_B, int EPI, int BM, int BN>
__global__ __launch_bounds__(128)
void gemm_wmma(const bf16* __restrict__ A, long sA, int lda,
               const bf16* __restrict__ B, long sB, int ldb,
               void* __restrict__ Cout, long sC, int ldc,
               int N, int K, float scale,
               const float* __restrict__ bias, float* __restrict__ hres,
               bf16* __restrict__ qbuf, bf16* __restrict__ kbuf, bf16* __restrict__ vbuf)
{
  const int wave = threadIdx.x >> 5;
  const int lane = threadIdx.x & 31;
  const int tmB  = blockIdx.x * BM;                 // first tile-row
  const int tnB  = (blockIdx.y * 4 + wave) * BN;    // first tile-col
  if (tnB * 16 >= N) return;                        // wave-uniform
  const int batch = blockIdx.z;
  const bf16* Ab = A + (long)batch * sA;
  const bf16* Bb = B + (long)batch * sB;
  const int l15 = lane & 15;
  const int kb  = (lane >> 4) << 3;                 // 0 or 8

  const v8f vzero = {0.f,0.f,0.f,0.f,0.f,0.f,0.f,0.f};
  v8f acc[BM][BN];
#pragma unroll
  for (int mi = 0; mi < BM; ++mi)
#pragma unroll
    for (int ni = 0; ni < BN; ++ni) acc[mi][ni] = vzero;

  if constexpr (MODE_B == 0) {
    for (int kk = 0; kk < K; kk += 32) {
      v16bf af[BM], bfv[BN];
#pragma unroll
      for (int mi = 0; mi < BM; ++mi) {
        const bf16* ap = Ab + (long)((tmB + mi) * 16 + l15) * lda + kk + kb;
        bf16x8 lo = *(const bf16x8*)(ap);
        bf16x8 hi = *(const bf16x8*)(ap + 16);
        af[mi] = __builtin_shufflevector(lo, hi, 0,1,2,3,4,5,6,7,8,9,10,11,12,13,14,15);
      }
#pragma unroll
      for (int ni = 0; ni < BN; ++ni) {
        const bf16* bp = Bb + (long)((tnB + ni) * 16 + l15) * ldb + kk + kb;
        bf16x8 lo = *(const bf16x8*)(bp);
        bf16x8 hi = *(const bf16x8*)(bp + 16);
        bfv[ni] = __builtin_shufflevector(lo, hi, 0,1,2,3,4,5,6,7,8,9,10,11,12,13,14,15);
      }
#pragma unroll
      for (int mi = 0; mi < BM; ++mi)
#pragma unroll
        for (int ni = 0; ni < BN; ++ni)
          acc[mi][ni] = __builtin_amdgcn_wmma_f32_16x16x32_bf16(
              false, af[mi], false, bfv[ni], (short)0, acc[mi][ni], false, false);
    }
  } else {
    // wave-private double-buffered LDS staging of B tiles (32 K x 16 N each)
    __shared__ bf16 smem[4 * BN * 2 * 512];         // 1KB per tile-buffer
    const unsigned sbase =
        (unsigned)(unsigned long long)(lds_bf16*)&smem[0];
    const unsigned wslice = sbase + (unsigned)(wave * BN * 2) * 1024u;

    auto stage = [&](int db, int kk2) {
#pragma unroll
      for (int ni = 0; ni < BN; ++ni) {
        const bf16* gp = Bb + (long)(kk2 + lane) * ldb + (tnB + ni) * 16;
        unsigned dst = wslice + (unsigned)((ni * 2 + db) * 1024) + (unsigned)lane * 32u;
        unsigned long long ga = (unsigned long long)(size_t)gp;
        asm volatile("global_load_async_to_lds_b128 %0, %1, off"
                     :: "v"(dst), "v"(ga) : "memory");
        asm volatile("global_load_async_to_lds_b128 %0, %1, off offset:16"
                     :: "v"(dst + 16u), "v"(ga) : "memory");
      }
    };

    stage(0, 0);
    int db = 0;
    for (int kk = 0; kk < K; kk += 32) {
      if (kk + 32 < K) {
        stage(db ^ 1, kk + 32);
        asm volatile("s_wait_asynccnt %0" :: "n"(2 * BN) : "memory");
      } else {
        asm volatile("s_wait_asynccnt 0" ::: "memory");
      }
      v16bf af[BM], bfv[BN];
#pragma unroll
      for (int mi = 0; mi < BM; ++mi) {
        const bf16* ap = Ab + (long)((tmB + mi) * 16 + l15) * lda + kk + kb;
        bf16x8 lo = *(const bf16x8*)(ap);
        bf16x8 hi = *(const bf16x8*)(ap + 16);
        af[mi] = __builtin_shufflevector(lo, hi, 0,1,2,3,4,5,6,7,8,9,10,11,12,13,14,15);
      }
#pragma unroll
      for (int ni = 0; ni < BN; ++ni) {
        const bf16* sl = &smem[((wave * BN + ni) * 2 + db) * 512];
#pragma unroll
        for (int e = 0; e < 16; ++e) {
          int kx = kb + (e < 8 ? e : e + 8);
          bfv[ni][e] = sl[kx * 16 + l15];
        }
      }
#pragma unroll
      for (int mi = 0; mi < BM; ++mi)
#pragma unroll
        for (int ni = 0; ni < BN; ++ni)
          acc[mi][ni] = __builtin_amdgcn_wmma_f32_16x16x32_bf16(
              false, af[mi], false, bfv[ni], (short)0, acc[mi][ni], false, false);
      db ^= 1;
    }
  }

#pragma unroll
  for (int mi = 0; mi < BM; ++mi)
#pragma unroll
    for (int ni = 0; ni < BN; ++ni)
#pragma unroll
      for (int e = 0; e < 8; ++e) {
        const int row = (tmB + mi) * 16 + e + ((lane >> 4) << 3);
        const int col = (tnB + ni) * 16 + l15;
        float val = acc[mi][ni][e] * scale;
        if (EPI == 0) {
          ((float*)Cout)[(long)batch * sC + (long)row * ldc + col] = val;
        } else if (EPI == 1) {
          ((bf16*)Cout)[(long)batch * sC + (long)row * ldc + col] = (bf16)val;
        } else if (EPI == 2) {
          const int which = col / CDIM;                 // 0=q 1=k 2=v
          const int ch = col - which * CDIM;
          const int hh = ch / DHD;
          const int dd = ch - hh * DHD;
          bf16* tgt = (which == 0) ? qbuf : ((which == 1) ? kbuf : vbuf);
          const float s2 = (which == 0) ? 0.125f : 1.0f; // DH^-0.5
          tgt[(((long)(batch * NH + hh)) * NP + row) * DHD + dd] = (bf16)(val * s2);
        } else {
          const int bb = row / NP;
          const int ip = row - bb * NP;
          if (ip >= PADF) {
            long hidx = ((long)bb * NSEQ + (ip - PADF)) * CDIM + col;
            hres[hidx] += val + bias[col];
          }
        }
      }
}

// ---------------------------- small kernels --------------------------------
__global__ void cvt_bf16_kernel(const float* __restrict__ in, bf16* __restrict__ out, long n) {
  long i = (long)blockIdx.x * 256 + threadIdx.x;
  if (i < n) out[i] = (bf16)in[i];
}

__global__ void build_h_kernel(const float* __restrict__ feat,
                               const float* __restrict__ cls,
                               float* __restrict__ h) {
  long i = (long)blockIdx.x * 256 + threadIdx.x;
  const long total = (long)B_ * NSEQ * CDIM;
  if (i >= total) return;
  int c = (int)(i % CDIM);
  long bt = i / CDIM;
  int t = (int)(bt % NSEQ);
  long b = bt / NSEQ;
  h[i] = (t == 0) ? cls[c] : feat[((long)b * NTOK + (t - 1)) * CDIM + c];
}

// LayerNorm row of h -> bf16 xln (padded layout [B, NP, C], first PADF rows zero)
__global__ void ln_pad_kernel(const float* __restrict__ h,
                              const float* __restrict__ g,
                              const float* __restrict__ be,
                              bf16* __restrict__ xln) {
  long row = blockIdx.x;                  // over B_*NP
  int b  = (int)(row / NP);
  int ip = (int)(row - (long)b * NP);
  bf16* out = xln + row * CDIM;
  int tid = threadIdx.x;                  // 128 threads
  if (ip < PADF) {
    for (int i = tid; i < CDIM; i += 128) out[i] = (bf16)0.0f;
    return;
  }
  const float* x = h + ((long)b * NSEQ + (ip - PADF)) * CDIM;
  __shared__ float red[128];
  float s = 0.f;
  for (int i = tid; i < CDIM; i += 128) s += x[i];
  red[tid] = s; __syncthreads();
  for (int st = 64; st > 0; st >>= 1) { if (tid < st) red[tid] += red[tid + st]; __syncthreads(); }
  float mean = red[0] * (1.0f / CDIM); __syncthreads();
  float v2 = 0.f;
  for (int i = tid; i < CDIM; i += 128) { float d = x[i] - mean; v2 += d * d; }
  red[tid] = v2; __syncthreads();
  for (int st = 64; st > 0; st >>= 1) { if (tid < st) red[tid] += red[tid + st]; __syncthreads(); }
  float inv = rsqrtf(red[0] * (1.0f / CDIM) + 1e-5f); __syncthreads();
  for (int i = tid; i < CDIM; i += 128)
    out[i] = (bf16)((x[i] - mean) * inv * g[i] + be[i]);
}

// landmark means: ql/kl[bh,i,d] = mean_{j<LSEG} q/k[bh, i*LSEG+j, d]
__global__ void landmark_kernel(const bf16* __restrict__ q, const bf16* __restrict__ k,
                                bf16* __restrict__ ql, bf16* __restrict__ kl) {
  int d = threadIdx.x;                    // 64
  long bm = blockIdx.x;                   // BH*MLAND
  long bh = bm >> 8;
  int i = (int)(bm & 255);
  const long base = (bh * NP + (long)i * LSEG) * DHD + d;
  float sq = 0.f, sk = 0.f;
#pragma unroll
  for (int j = 0; j < LSEG; ++j) {
    sq += (float)q[base + (long)j * DHD];
    sk += (float)k[base + (long)j * DHD];
  }
  const float inv = 1.0f / (float)LSEG;
  ql[(bh * MLAND + i) * DHD + d] = (bf16)(sq * inv);
  kl[(bh * MLAND + i) * DHD + d] = (bf16)(sk * inv);
}

// in-place row softmax on a bf16 matrix (fp32 math)
__global__ void softmax_kernel(bf16* __restrict__ buf, int rowlen) {
  long row = blockIdx.x;
  bf16* p = buf + row * (long)rowlen;
  __shared__ float red[256];
  int tid = threadIdx.x;
  float mx = -3.0e38f;
  for (int i = tid; i < rowlen; i += 256) mx = fmaxf(mx, (float)p[i]);
  red[tid] = mx; __syncthreads();
  for (int s = 128; s > 0; s >>= 1) { if (tid < s) red[tid] = fmaxf(red[tid], red[tid + s]); __syncthreads(); }
  mx = red[0]; __syncthreads();
  float sum = 0.f;
  for (int i = tid; i < rowlen; i += 256) sum += __expf((float)p[i] - mx);
  red[tid] = sum; __syncthreads();
  for (int s = 128; s > 0; s >>= 1) { if (tid < s) red[tid] += red[tid + s]; __syncthreads(); }
  float inv = 1.0f / red[0]; __syncthreads();
  for (int i = tid; i < rowlen; i += 256) p[i] = (bf16)(__expf((float)p[i] - mx) * inv);
}

// global max column-sum of a2 (rows are softmax -> row-sums == 1)
__global__ void colmax_kernel(const bf16* __restrict__ a2, float* __restrict__ scal) {
  int bh = blockIdx.x;                    // BH
  int j = threadIdx.x;                    // 256 columns
  const bf16* p = a2 + (long)bh * MLAND * MLAND + j;
  float s = 0.f;
  for (int i = 0; i < MLAND; ++i) s += (float)p[(long)i * MLAND];
  __shared__ float red[256];
  red[j] = s; __syncthreads();
  for (int st = 128; st > 0; st >>= 1) { if (j < st) red[j] = fmaxf(red[j], red[j + st]); __syncthreads(); }
  if (j == 0) atomicMax((unsigned*)scal, __float_as_uint(red[0]));
}

// z0 = a2^T / colmax
__global__ void tscale_kernel(const bf16* __restrict__ a2, const float* __restrict__ scal,
                              bf16* __restrict__ z) {
  long idx = (long)blockIdx.x * 256 + threadIdx.x;
  if (idx >= (long)BH * MLAND * MLAND) return;
  int r  = (int)((idx >> 8) & 255);
  int ct = (int)(idx & 255);
  float s = 1.0f / (*scal);
  z[idx] = (bf16)((float)a2[(idx & ~(long)65535) + ((long)ct << 8) + r] * s);
}

// out = alpha*I - x   (in-place safe)
__global__ void eyeminus_kernel(const bf16* __restrict__ x, bf16* __restrict__ out, float alpha) {
  long idx = (long)blockIdx.x * 256 + threadIdx.x;
  if (idx >= (long)BH * MLAND * MLAND) return;
  int i = (int)((idx >> 8) & 255);
  int j = (int)(idx & 255);
  out[idx] = (bf16)(((i == j) ? alpha : 0.f) - (float)x[idx]);
}

// depthwise 33-tap conv along sequence on v, accumulated into out_head (f32)
__global__ void resconv_kernel(const bf16* __restrict__ v, const float* __restrict__ w,
                               float* __restrict__ oh) {
  int d = threadIdx.x;                    // 64
  long idx = blockIdx.x;                  // BH*NP
  long bh = idx / NP;
  int nn = (int)(idx - bh * NP);
  int hh = (int)(bh % NH);
  float acc = 0.f;
#pragma unroll
  for (int t = 0; t < RESK; ++t) {
    int s = nn + t - RESK / 2;
    if (s >= 0 && s < NP) acc += (float)v[(bh * NP + s) * DHD + d] * w[hh * RESK + t];
  }
  oh[(bh * NP + nn) * DHD + d] += acc;
}

// [BH, NP, DHD] f32 -> [B, NP, C] bf16 (heads back together)
__global__ void combine_kernel(const float* __restrict__ oh, bf16* __restrict__ outc) {
  long idx = (long)blockIdx.x * 256 + threadIdx.x;
  const long total = (long)B_ * NP * CDIM;
  if (idx >= total) return;
  int c = (int)(idx % CDIM);
  long bn = idx / CDIM;
  int ip = (int)(bn % NP);
  long b = bn / NP;
  int hh = c / DHD, dd = c - hh * DHD;
  outc[idx] = (bf16)oh[((b * NH + hh) * (long)NP + ip) * DHD + dd];
}

// tokens 1..4096 of h -> NCHW f32 buffer
__global__ void tok2cnn_kernel(const float* __restrict__ h, float* __restrict__ cnn) {
  long idx = (long)blockIdx.x * 256 + threadIdx.x;
  const long total = (long)B_ * CDIM * HHW * HHW;
  if (idx >= total) return;
  int p = (int)(idx & 4095);
  long bc = idx >> 12;
  int c = (int)(bc % CDIM);
  long b = bc / CDIM;
  cnn[idx] = h[((long)b * NSEQ + 1 + p) * CDIM + c];
}

// y = x + dw7(x)+b7 + dw5(x)+b5 + dw3(x)+b3 ; written back into h tokens
__global__ void cnn_kernel(const float* __restrict__ cnn,
                           const float* __restrict__ w7, const float* __restrict__ b7,
                           const float* __restrict__ w5, const float* __restrict__ b5,
                           const float* __restrict__ w3, const float* __restrict__ b3,
                           float* __restrict__ h) {
  long idx = (long)blockIdx.x * 256 + threadIdx.x;
  const long total = (long)B_ * CDIM * HHW * HHW;
  if (idx >= total) return;
  int j = (int)(idx & 63);
  int i = (int)((idx >> 6) & 63);
  long bc = idx >> 12;
  int c = (int)(bc % CDIM);
  long b = bc / CDIM;
  const float* base = cnn + bc * HHW * HHW;
  float acc = base[i * HHW + j] + b7[c] + b5[c] + b3[c];
  for (int dy = -3; dy <= 3; ++dy)
    for (int dx = -3; dx <= 3; ++dx) {
      int y = i + dy, x = j + dx;
      if ((unsigned)y < HHW && (unsigned)x < HHW)
        acc += base[y * HHW + x] * w7[c * 49 + (dy + 3) * 7 + (dx + 3)];
    }
  for (int dy = -2; dy <= 2; ++dy)
    for (int dx = -2; dx <= 2; ++dx) {
      int y = i + dy, x = j + dx;
      if ((unsigned)y < HHW && (unsigned)x < HHW)
        acc += base[y * HHW + x] * w5[c * 25 + (dy + 2) * 5 + (dx + 2)];
    }
  for (int dy = -1; dy <= 1; ++dy)
    for (int dx = -1; dx <= 1; ++dx) {
      int y = i + dy, x = j + dx;
      if ((unsigned)y < HHW && (unsigned)x < HHW)
        acc += base[y * HHW + x] * w3[c * 9 + (dy + 1) * 3 + (dx + 1)];
    }
  h[((long)b * NSEQ + 1 + i * HHW + j) * CDIM + c] = acc;
}

// d_out = concat(h[:,0] , h[:,1:]) flat f32
__global__ void writeout_kernel(const float* __restrict__ h, float* __restrict__ out) {
  long idx = (long)blockIdx.x * 256 + threadIdx.x;
  const long ncls = (long)B_ * CDIM;
  const long total = ncls + (long)B_ * NTOK * CDIM;
  if (idx >= total) return;
  if (idx < ncls) {
    long b = idx / CDIM; int c = (int)(idx % CDIM);
    out[idx] = h[b * (long)NSEQ * CDIM + c];
  } else {
    long j = idx - ncls;
    int c = (int)(j % CDIM);
    long bt = j / CDIM;
    long t = bt % NTOK;
    long b = bt / NTOK;
    out[idx] = h[((long)b * NSEQ + 1 + t) * CDIM + c];
  }
}

// ---------------------------------------------------------------------------
extern "C" void kernel_launch(void* const* d_in, const int* in_sizes, int n_in,
                              void* d_out, int out_size, void* d_ws, size_t ws_size,
                              hipStream_t stream) {
  (void)in_sizes; (void)n_in; (void)out_size; (void)ws_size;
  const float* features = (const float*)d_in[0];
  const float* cls      = (const float*)d_in[1];
  const float* ln1_g = (const float*)d_in[2];
  const float* ln1_b = (const float*)d_in[3];
  const float* qkv1_w = (const float*)d_in[4];
  const float* out1_w = (const float*)d_in[5];
  const float* out1_b = (const float*)d_in[6];
  const float* res1_w = (const float*)d_in[7];
  const float* p7_w = (const float*)d_in[8];
  const float* p7_b = (const float*)d_in[9];
  const float* p5_w = (const float*)d_in[10];
  const float* p5_b = (const float*)d_in[11];
  const float* p3_w = (const float*)d_in[12];
  const float* p3_b = (const float*)d_in[13];
  const float* ln2_g = (const float*)d_in[14];
  const float* ln2_b = (const float*)d_in[15];
  const float* qkv2_w = (const float*)d_in[16];
  const float* out2_w = (const float*)d_in[17];
  const float* out2_b = (const float*)d_in[18];
  const float* res2_w = (const float*)d_in[19];

  char* base = (char*)d_ws;
  size_t cur = 0;
  auto alloc = [&](size_t bytes) -> void* {
    cur = (cur + 255) & ~(size_t)255;
    void* p = base + cur;
    cur += bytes;
    return p;
  };

  float* h     = (float*)alloc((size_t)B_ * NSEQ * CDIM * 4);
  bf16* xln    = (bf16*)alloc((size_t)B_ * NP * CDIM * 2);
  bf16* q      = (bf16*)alloc((size_t)BH * NP * DHD * 2);
  bf16* k      = (bf16*)alloc((size_t)BH * NP * DHD * 2);
  bf16* v      = (bf16*)alloc((size_t)BH * NP * DHD * 2);
  bf16* ql     = (bf16*)alloc((size_t)BH * MLAND * DHD * 2);
  bf16* kl     = (bf16*)alloc((size_t)BH * MLAND * DHD * 2);
  bf16* a1     = (bf16*)alloc((size_t)BH * NP * MLAND * 2);
  bf16* a2     = (bf16*)alloc((size_t)BH * MLAND * MLAND * 2);
  bf16* a3     = (bf16*)alloc((size_t)BH * MLAND * NP * 2);
  bf16* zA     = (bf16*)alloc((size_t)BH * MLAND * MLAND * 2);
  bf16* zB     = (bf16*)alloc((size_t)BH * MLAND * MLAND * 2);
  bf16* xz     = (bf16*)alloc((size_t)BH * MLAND * MLAND * 2);
  bf16* t1     = (bf16*)alloc((size_t)BH * MLAND * MLAND * 2);
  bf16* t2     = (bf16*)alloc((size_t)BH * MLAND * MLAND * 2);
  bf16* a3v    = (bf16*)alloc((size_t)BH * MLAND * DHD * 2);
  bf16* tmp2   = (bf16*)alloc((size_t)BH * MLAND * DHD * 2);
  float* ohead = (float*)alloc((size_t)BH * NP * DHD * 4);
  bf16* outc   = (bf16*)alloc((size_t)B_ * NP * CDIM * 2);
  float* cnn   = (float*)alloc((size_t)B_ * CDIM * HHW * HHW * 4);
  bf16* wq0    = (bf16*)alloc((size_t)3 * CDIM * CDIM * 2);
  bf16* wo0    = (bf16*)alloc((size_t)CDIM * CDIM * 2);
  bf16* wq1    = (bf16*)alloc((size_t)3 * CDIM * CDIM * 2);
  bf16* wo1    = (bf16*)alloc((size_t)CDIM * CDIM * 2);
  float* scal  = (float*)alloc(256);

  auto cdiv = [](long a, long b) -> unsigned { return (unsigned)((a + b - 1) / b); };
  const long MM = (long)BH * MLAND * MLAND;

  cvt_bf16_kernel<<<cdiv((long)3 * CDIM * CDIM, 256), 256, 0, stream>>>(qkv1_w, wq0, (long)3 * CDIM * CDIM);
  cvt_bf16_kernel<<<cdiv((long)CDIM * CDIM, 256), 256, 0, stream>>>(out1_w, wo0, (long)CDIM * CDIM);
  cvt_bf16_kernel<<<cdiv((long)3 * CDIM * CDIM, 256), 256, 0, stream>>>(qkv2_w, wq1, (long)3 * CDIM * CDIM);
  cvt_bf16_kernel<<<cdiv((long)CDIM * CDIM, 256), 256, 0, stream>>>(out2_w, wo1, (long)CDIM * CDIM);

  build_h_kernel<<<cdiv((long)B_ * NSEQ * CDIM, 256), 256, 0, stream>>>(features, cls, h);

  auto attn = [&](const float* lng, const float* lnb, bf16* wqkv, bf16* wout,
                  const float* outb, const float* resw) {
    ln_pad_kernel<<<B_ * NP, 128, 0, stream>>>(h, lng, lnb, xln);
    // QKV: [B, NP, C] x Wqkv^T -> per-head q(0.125x)/k/v bf16 buffers
    gemm_wmma<0, 2, 2, 2><<<dim3(NP / 32, (3 * CDIM) / 128, B_), 128, 0, stream>>>(
        xln, (long)NP * CDIM, CDIM, wqkv, 0, CDIM,
        nullptr, 0, 0, 3 * CDIM, CDIM, 1.f, nullptr, nullptr, q, k, v);
    landmark_kernel<<<BH * MLAND, 64, 0, stream>>>(q, k, ql, kl);
    // sim1 = q @ kl^T -> softmax -> a1
    gemm_wmma<0, 1, 2, 2><<<dim3(NP / 32, MLAND / 128, BH), 128, 0, stream>>>(
        q, (long)NP * DHD, DHD, kl, (long)MLAND * DHD, DHD,
        a1, (long)NP * MLAND, MLAND, MLAND, DHD, 1.f, nullptr, nullptr, nullptr, nullptr, nullptr);
    softmax_kernel<<<BH * NP, 256, 0, stream>>>(a1, MLAND);
    // sim2 = ql @ kl^T -> softmax -> a2
    gemm_wmma<0, 1, 2, 2><<<dim3(MLAND / 32, MLAND / 128, BH), 128, 0, stream>>>(
        ql, (long)MLAND * DHD, DHD, kl, (long)MLAND * DHD, DHD,
        a2, (long)MLAND * MLAND, MLAND, MLAND, DHD, 1.f, nullptr, nullptr, nullptr, nullptr, nullptr);
    softmax_kernel<<<BH * MLAND, 256, 0, stream>>>(a2, MLAND);
    // Moore-Penrose pinv init: row-sums==1 (softmax), so z0 = a2^T / max-col-sum
    hipMemsetAsync(scal, 0, 4, stream);
    colmax_kernel<<<BH, 256, 0, stream>>>(a2, scal);
    tscale_kernel<<<cdiv(MM, 256), 256, 0, stream>>>(a2, scal, zA);
    bf16* zc = zA; bf16* zn = zB;
    for (int it = 0; it < 6; ++it) {
      gemm_wmma<1, 1, 2, 2><<<dim3(MLAND / 32, MLAND / 128, BH), 128, 0, stream>>>(
          a2, MM / BH, MLAND, zc, MM / BH, MLAND,
          xz, MM / BH, MLAND, MLAND, MLAND, 1.f, nullptr, nullptr, nullptr, nullptr, nullptr);
      eyeminus_kernel<<<cdiv(MM, 256), 256, 0, stream>>>(xz, t1, 7.f);
      gemm_wmma<1, 1, 2, 2><<<dim3(MLAND / 32, MLAND / 128, BH), 128, 0, stream>>>(
          xz, MM / BH, MLAND, t1, MM / BH, MLAND,
          t2, MM / BH, MLAND, MLAND, MLAND, 1.f, nullptr, nullptr, nullptr, nullptr, nullptr);
      eyeminus_kernel<<<cdiv(MM, 256), 256, 0, stream>>>(t2, t2, 15.f);
      gemm_wmma<1, 1, 2, 2><<<dim3(MLAND / 32, MLAND / 128, BH), 128, 0, stream>>>(
          xz, MM / BH, MLAND, t2, MM / BH, MLAND,
          t1, MM / BH, MLAND, MLAND, MLAND, 1.f, nullptr, nullptr, nullptr, nullptr, nullptr);
      eyeminus_kernel<<<cdiv(MM, 256), 256, 0, stream>>>(t1, t1, 13.f);
      gemm_wmma<1, 1, 2, 2><<<dim3(MLAND / 32, MLAND / 128, BH), 128, 0, stream>>>(
          zc, MM / BH, MLAND, t1, MM / BH, MLAND,
          zn, MM / BH, MLAND, MLAND, MLAND, 0.25f, nullptr, nullptr, nullptr, nullptr, nullptr);
      bf16* tsw = zc; zc = zn; zn = tsw;
    }
    // sim3 = ql @ k^T -> softmax -> a3
    gemm_wmma<0, 1, 2, 2><<<dim3(MLAND / 32, NP / 128, BH), 128, 0, stream>>>(
        ql, (long)MLAND * DHD, DHD, k, (long)NP * DHD, DHD,
        a3, (long)MLAND * NP, NP, NP, DHD, 1.f, nullptr, nullptr, nullptr, nullptr, nullptr);
    softmax_kernel<<<BH * MLAND, 256, 0, stream>>>(a3, NP);
    // right-associated chain: a3v = a3@v ; tmp2 = z@a3v ; ohead = a1@tmp2
    gemm_wmma<1, 1, 2, 1><<<dim3(MLAND / 32, 1, BH), 128, 0, stream>>>(
        a3, (long)MLAND * NP, NP, v, (long)NP * DHD, DHD,
        a3v, (long)MLAND * DHD, DHD, DHD, NP, 1.f, nullptr, nullptr, nullptr, nullptr, nullptr);
    gemm_wmma<1, 1, 2, 1><<<dim3(MLAND / 32, 1, BH), 128, 0, stream>>>(
        zc, MM / BH, MLAND, a3v, (long)MLAND * DHD, DHD,
        tmp2, (long)MLAND * DHD, DHD, DHD, MLAND, 1.f, nullptr, nullptr, nullptr, nullptr, nullptr);
    gemm_wmma<1, 0, 2, 1><<<dim3(NP / 32, 1, BH), 128, 0, stream>>>(
        a1, (long)NP * MLAND, MLAND, tmp2, (long)MLAND * DHD, DHD,
        ohead, (long)NP * DHD, DHD, DHD, MLAND, 1.f, nullptr, nullptr, nullptr, nullptr, nullptr);
    resconv_kernel<<<BH * NP, 64, 0, stream>>>(v, resw, ohead);
    combine_kernel<<<cdiv((long)B_ * NP * CDIM, 256), 256, 0, stream>>>(ohead, outc);
    // out-proj + bias, drop PADF front rows, residual-add into h
    gemm_wmma<0, 3, 2, 2><<<dim3((B_ * NP) / 32, CDIM / 128, 1), 128, 0, stream>>>(
        outc, 0, CDIM, wout, 0, CDIM,
        nullptr, 0, CDIM, CDIM, CDIM, 1.f, outb, h, nullptr, nullptr, nullptr);
  };

  attn(ln1_g, ln1_b, wq0, wo0, out1_b, res1_w);

  tok2cnn_kernel<<<cdiv((long)B_ * CDIM * HHW * HHW, 256), 256, 0, stream>>>(h, cnn);
  cnn_kernel<<<cdiv((long)B_ * CDIM * HHW * HHW, 256), 256, 0, stream>>>(
      cnn, p7_w, p7_b, p5_w, p5_b, p3_w, p3_b, h);

  attn(ln2_g, ln2_b, wq1, wo1, out2_b, res2_w);

  writeout_kernel<<<cdiv((long)B_ * CDIM + (long)B_ * NTOK * CDIM, 256), 256, 0, stream>>>(
      h, (float*)d_out);
}
